// Block_37512244363306
// MI455X (gfx1250) — compile-verified, matrix operand order checked by reading, not run on hardware
//
#include <hip/hip_runtime.h>
#include <math.h>

// ---------------------------------------------------------------- constants
#define BB      4
#define CC      16
#define TT      262144
#define WIN     512
#define HOPS    256
#define NF      1024          // frames
#define NCOEF   257           // rfft bins
#define NFWD    514           // re(257) + im(257)
#define NPAD_F  576           // forward N padded to multiple of 64
#define KINV    544           // inverse K padded to multiple of 32
#define TPAD    (TT + HOPS)   // zero-padded mixed signal length

#define LDS_STR_F (WIN + 8)   // 520 halfs: bank-conflict-free b128 pattern
#define LDS_STR_I (KINV + 8)  // 552 halfs

#define AS1 __attribute__((address_space(1)))
#define AS3 __attribute__((address_space(3)))

#if __has_builtin(__builtin_amdgcn_global_load_async_to_lds_b128) && \
    __has_builtin(__builtin_amdgcn_s_wait_asynccnt)
#define HAVE_ASYNC_LDS 1
#else
#define HAVE_ASYNC_LDS 0
#endif

typedef _Float16 v16h __attribute__((ext_vector_type(16)));
typedef _Float16 v8h  __attribute__((ext_vector_type(8)));
typedef float    v8f  __attribute__((ext_vector_type(8)));
typedef int      v4i  __attribute__((ext_vector_type(4)));

typedef AS1 v4i* gv4i_ptr;   // global-AS 16-byte chunk pointer
typedef AS3 v4i* lv4i_ptr;   // LDS-AS 16-byte chunk pointer

struct HPair { v8h lo, hi; };
union  HFrag { v16h v; HPair h; };

// ------------------------------------------------- 1) channel mix (mem-bound)
__global__ void mix_kernel(const float* __restrict__ x,
                           const float* __restrict__ M,
                           _Float16* __restrict__ mixed) {
  long tid = (long)blockIdx.x * blockDim.x + threadIdx.x;   // over BB*TPAD
  if (tid >= (long)BB * TPAD) return;
  int b = (int)(tid / TPAD);
  int t = (int)(tid % TPAD);
  float xin[CC];
  if (t < TT) {
    #pragma unroll
    for (int c = 0; c < CC; ++c) xin[c] = x[((long)(b * CC + c)) * TT + t];
  } else {
    #pragma unroll
    for (int c = 0; c < CC; ++c) xin[c] = 0.f;
  }
  #pragma unroll
  for (int d = 0; d < CC; ++d) {
    float acc = 0.f;
    #pragma unroll
    for (int c = 0; c < CC; ++c) acc += xin[c] * M[c * CC + d];
    mixed[((long)(b * CC + d)) * TPAD + t] = (_Float16)acc;
  }
}

// ------------------------------------------------- 2) DFT matrix builders
// Forward B, stored n-major: Bf[n][k], n<257 -> cos, 257..513 -> -sin, pad 0.
__global__ void build_fwd(_Float16* __restrict__ Bf) {
  int tid = blockIdx.x * blockDim.x + threadIdx.x;          // NPAD_F * WIN
  if (tid >= NPAD_F * WIN) return;
  int n = tid / WIN, k = tid % WIN;
  const float w0 = 6.283185307179586f / (float)WIN;
  float v = 0.f;
  if (n < NCOEF)      v =  __cosf(w0 * (float)((n * k) & (WIN - 1)));
  else if (n < NFWD)  v = -__sinf(w0 * (float)(((n - NCOEF) * k) & (WIN - 1)));
  Bf[tid] = (_Float16)v;
}

// Inverse B, stored s-major: Bi[s][k].  Folds Hermitian weights, 1/N scale and
// the Hann OLA window into the matrix so the epilogue is a pure 2-tap gather.
__global__ void build_inv(_Float16* __restrict__ Bi) {
  int tid = blockIdx.x * blockDim.x + threadIdx.x;          // WIN * KINV
  if (tid >= WIN * KINV) return;
  int s = tid / KINV, k = tid % KINV;
  const float w0 = 6.283185307179586f / (float)WIN;
  float hann = 0.5f - 0.5f * __cosf(w0 * (float)s);
  float v = 0.f;
  if (k < NCOEF) {
    float w = (k == 0 || k == WIN / 2) ? 1.f : 2.f;
    v =  w * __cosf(w0 * (float)((k * s) & (WIN - 1)));
  } else if (k < NFWD) {
    int kk = k - NCOEF;
    float w = (kk == 0 || kk == WIN / 2) ? 0.f : 2.f;      // irfft ignores Im[0],Im[N/2]
    v = -w * __sinf(w0 * (float)((kk * s) & (WIN - 1)));
  }
  Bi[tid] = (_Float16)(v * hann * (1.f / (float)WIN));
}

// ---------------------------------------------------------------- helpers
// Stage 64 rows of K halfs (row-major, LDS row stride = K+8) into LDS.
// 128 threads; 2 threads per row, each copies K/16 b128 chunks.
template <int K, int STR>
__device__ __forceinline__ void stage_B(const _Float16* __restrict__ gB,
                                        _Float16* lds, int tid) {
  const int col = tid >> 1;
  const int h   = tid & 1;
  #pragma unroll
  for (int i = 0; i < K / 16; ++i) {
    int pos = (h * (K / 16) + i) * 8;
#if HAVE_ASYNC_LDS
    __builtin_amdgcn_global_load_async_to_lds_b128(
        (gv4i_ptr)(gB + (size_t)col * K + pos),
        (lv4i_ptr)&lds[col * STR + pos], 0, 0);
#else
    *(v8h*)&lds[col * STR + pos] = *(const v8h*)(gB + (size_t)col * K + pos);
#endif
  }
#if HAVE_ASYNC_LDS
  __builtin_amdgcn_s_wait_asynccnt(0);
#endif
  __syncthreads();
}

// ------------------------------------------------- 3) forward DFT via WMMA
// Block = 4 waves x (1 row-tile each), sharing B for 4 n-tiles from LDS.
// Per k-step per wave: 1 A-frag (2x global b128) feeds 4 WMMAs.
__global__ void __launch_bounds__(128)
fwd_gemm(const _Float16* __restrict__ mixed,
         const _Float16* __restrict__ Bf,
         _Float16* __restrict__ spec) {
  __shared__ _Float16 ldsB[64 * LDS_STR_F];
  const int tid = threadIdx.x;
  const int ngroup   = blockIdx.x % (NPAD_F / 64);          // 9 groups
  const int rowGroup = blockIdx.x / (NPAD_F / 64);

  stage_B<WIN, LDS_STR_F>(Bf + (size_t)ngroup * 64 * WIN, ldsB, tid);

  const int wv = tid >> 5, lane = tid & 31;
  const int rowTile = rowGroup * 4 + wv;
  const int bc = rowTile >> 6;                              // / (NF/16)
  const int ft = rowTile & 63;
  const int frameBase = ft * 16;
  const int m = lane & 15, half = lane >> 4;
  const int kb0 = half * 8;
  const _Float16* arow = mixed + (size_t)bc * TPAD + (size_t)(frameBase + m) * HOPS;
  const _Float16* lb   = &ldsB[m * LDS_STR_F];

  v8f acc0 = {}, acc1 = {}, acc2 = {}, acc3 = {};
  #pragma unroll 2
  for (int ks = 0; ks < WIN / 32; ++ks) {                   // 16 k-steps
    int kb = kb0 + ks * 32;
    HFrag a, b0, b1, b2, b3;
    a.h.lo  = *(const v8h*)(arow + kb);
    a.h.hi  = *(const v8h*)(arow + kb + 16);
    b0.h.lo = *(const v8h*)(lb + 0 * 16 * LDS_STR_F + kb);
    b0.h.hi = *(const v8h*)(lb + 0 * 16 * LDS_STR_F + kb + 16);
    b1.h.lo = *(const v8h*)(lb + 1 * 16 * LDS_STR_F + kb);
    b1.h.hi = *(const v8h*)(lb + 1 * 16 * LDS_STR_F + kb + 16);
    b2.h.lo = *(const v8h*)(lb + 2 * 16 * LDS_STR_F + kb);
    b2.h.hi = *(const v8h*)(lb + 2 * 16 * LDS_STR_F + kb + 16);
    b3.h.lo = *(const v8h*)(lb + 3 * 16 * LDS_STR_F + kb);
    b3.h.hi = *(const v8h*)(lb + 3 * 16 * LDS_STR_F + kb + 16);
    acc0 = __builtin_amdgcn_wmma_f32_16x16x32_f16(false, a.v, false, b0.v, (short)0, acc0, false, false);
    acc1 = __builtin_amdgcn_wmma_f32_16x16x32_f16(false, a.v, false, b1.v, (short)0, acc1, false, false);
    acc2 = __builtin_amdgcn_wmma_f32_16x16x32_f16(false, a.v, false, b2.v, (short)0, acc2, false, false);
    acc3 = __builtin_amdgcn_wmma_f32_16x16x32_f16(false, a.v, false, b3.v, (short)0, acc3, false, false);
  }

  const long rowBase = ((long)bc * NF + frameBase + 8 * half) * NPAD_F;
  const int  nbase   = ngroup * 64 + m;
  #pragma unroll
  for (int r = 0; r < 8; ++r) {
    long o = rowBase + (long)r * NPAD_F + nbase;
    spec[o +  0] = (_Float16)acc0[r];
    spec[o + 16] = (_Float16)acc1[r];
    spec[o + 32] = (_Float16)acc2[r];
    spec[o + 48] = (_Float16)acc3[r];
  }
}

// ------------------------------------------------- 4) per-frame recurrence
// y_i = (x_i + y_{i-1}) * H  (H real), 1 thread per (b,c,bin,re/im) chain.
__global__ void scan_kernel(const _Float16* __restrict__ spec,
                            const float* __restrict__ transfer,
                            _Float16* __restrict__ yspec) {
  int tid = blockIdx.x * blockDim.x + threadIdx.x;          // BB*CC*KINV
  if (tid >= BB * CC * KINV) return;
  int bc = tid / KINV;
  int n  = tid % KINV;
  int c  = bc & (CC - 1);
  bool valid = n < NFWD;
  int k  = (n < NCOEF) ? n : (valid ? n - NCOEF : 0);
  float h = valid ? transfer[c * NCOEF + k] : 0.f;
  const _Float16* sp = spec  + (long)bc * NF * NPAD_F + (valid ? n : 0);
  _Float16*       yp = yspec + (long)bc * NF * KINV   + n;
  float y = 0.f;
  for (int f = 0; f < NF; ++f) {
    float xv = valid ? (float)sp[(long)f * NPAD_F] : 0.f;
    y = (xv + y) * h;                                       // h==0 for pad cols
    yp[(long)f * KINV] = (_Float16)y;
  }
}

// ------------------------------------------------- 5) inverse DFT via WMMA
__global__ void __launch_bounds__(128)
inv_gemm(const _Float16* __restrict__ yspec,
         const _Float16* __restrict__ Bi,
         _Float16* __restrict__ fw) {
  __shared__ _Float16 ldsB[64 * LDS_STR_I];
  const int tid = threadIdx.x;
  const int ngroup   = blockIdx.x & 7;                      // WIN/64 = 8 groups
  const int rowGroup = blockIdx.x >> 3;

  stage_B<KINV, LDS_STR_I>(Bi + (size_t)ngroup * 64 * KINV, ldsB, tid);

  const int wv = tid >> 5, lane = tid & 31;
  const int rowTile = rowGroup * 4 + wv;
  const int bc = rowTile >> 6;
  const int ft = rowTile & 63;
  const int frameBase = ft * 16;
  const int m = lane & 15, half = lane >> 4;
  const int kb0 = half * 8;
  const _Float16* arow = yspec + ((long)bc * NF + frameBase + m) * KINV;
  const _Float16* lb   = &ldsB[m * LDS_STR_I];

  v8f acc0 = {}, acc1 = {}, acc2 = {}, acc3 = {};
  #pragma unroll 2
  for (int ks = 0; ks < KINV / 32; ++ks) {                  // 17 k-steps
    int kb = kb0 + ks * 32;
    HFrag a, b0, b1, b2, b3;
    a.h.lo  = *(const v8h*)(arow + kb);
    a.h.hi  = *(const v8h*)(arow + kb + 16);
    b0.h.lo = *(const v8h*)(lb + 0 * 16 * LDS_STR_I + kb);
    b0.h.hi = *(const v8h*)(lb + 0 * 16 * LDS_STR_I + kb + 16);
    b1.h.lo = *(const v8h*)(lb + 1 * 16 * LDS_STR_I + kb);
    b1.h.hi = *(const v8h*)(lb + 1 * 16 * LDS_STR_I + kb + 16);
    b2.h.lo = *(const v8h*)(lb + 2 * 16 * LDS_STR_I + kb);
    b2.h.hi = *(const v8h*)(lb + 2 * 16 * LDS_STR_I + kb + 16);
    b3.h.lo = *(const v8h*)(lb + 3 * 16 * LDS_STR_I + kb);
    b3.h.hi = *(const v8h*)(lb + 3 * 16 * LDS_STR_I + kb + 16);
    acc0 = __builtin_amdgcn_wmma_f32_16x16x32_f16(false, a.v, false, b0.v, (short)0, acc0, false, false);
    acc1 = __builtin_amdgcn_wmma_f32_16x16x32_f16(false, a.v, false, b1.v, (short)0, acc1, false, false);
    acc2 = __builtin_amdgcn_wmma_f32_16x16x32_f16(false, a.v, false, b2.v, (short)0, acc2, false, false);
    acc3 = __builtin_amdgcn_wmma_f32_16x16x32_f16(false, a.v, false, b3.v, (short)0, acc3, false, false);
  }

  const long rowBase = ((long)bc * NF + frameBase + 8 * half) * WIN;
  const int  nbase   = ngroup * 64 + m;
  #pragma unroll
  for (int r = 0; r < 8; ++r) {
    long o = rowBase + (long)r * WIN + nbase;
    fw[o +  0] = (_Float16)acc0[r];
    fw[o + 16] = (_Float16)acc1[r];
    fw[o + 32] = (_Float16)acc2[r];
    fw[o + 48] = (_Float16)acc3[r];
  }
}

// ------------------------------------------------- 6) OLA (2-tap) + tanh
__global__ void ola_kernel(const _Float16* __restrict__ fw,
                           const float* __restrict__ gain,
                           float* __restrict__ out) {
  long tid = (long)blockIdx.x * blockDim.x + threadIdx.x;   // BB*CC*TT
  if (tid >= (long)BB * CC * TT) return;
  long bc = tid / TT;
  int  t  = (int)(tid % TT);
  int f0 = t >> 8;
  int s0 = t & 255;
  const _Float16* fp = fw + bc * (long)NF * WIN;
  float v = (float)fp[(long)f0 * WIN + s0];
  if (f0 > 0) v += (float)fp[(long)(f0 - 1) * WIN + s0 + HOPS];
  out[tid] = tanhf(v * gain[0]);
}

// ---------------------------------------------------------------- launcher
extern "C" void kernel_launch(void* const* d_in, const int* in_sizes, int n_in,
                              void* d_out, int out_size, void* d_ws, size_t ws_size,
                              hipStream_t stream) {
  const float* x        = (const float*)d_in[0];
  const float* M        = (const float*)d_in[1];
  const float* transfer = (const float*)d_in[2];
  const float* gain     = (const float*)d_in[3];
  float* out = (float*)d_out;

  char* ws = (char*)d_ws;
  size_t off = 0;
  auto alloc = [&](size_t bytes) -> void* {
    void* p = ws + off;
    off += (bytes + 255) & ~(size_t)255;
    return p;
  };
  _Float16* mixed = (_Float16*)alloc((size_t)BB * CC * TPAD * 2);        // 33.6 MB
  _Float16* Bf    = (_Float16*)alloc((size_t)NPAD_F * WIN * 2);          // 0.59 MB
  _Float16* Bi    = (_Float16*)alloc((size_t)WIN * KINV * 2);            // 0.56 MB
  _Float16* spec  = (_Float16*)alloc((size_t)BB * CC * NF * NPAD_F * 2); // 75.5 MB
  _Float16* yspec = (_Float16*)alloc((size_t)BB * CC * NF * KINV * 2);   // 71.3 MB
  _Float16* fw    = spec;   // spec is dead after scan; reuse (67.1 MB <= 75.5 MB)

  // 1) channel mix -> f16 mixed signal (zero-padded tail)
  {
    long n = (long)BB * TPAD;
    mix_kernel<<<(unsigned)((n + 255) / 256), 256, 0, stream>>>(x, M, mixed);
  }
  // 2) DFT matrices (L2-resident)
  build_fwd<<<(NPAD_F * WIN + 255) / 256, 256, 0, stream>>>(Bf);
  build_inv<<<(WIN * KINV + 255) / 256, 256, 0, stream>>>(Bi);
  // 3) forward DFT GEMM: 1024 row-groups x 9 n-groups, 4 waves/block
  {
    int blocks = (BB * CC * NF / 64) * (NPAD_F / 64);  // 9216
    fwd_gemm<<<blocks, 128, 0, stream>>>(mixed, Bf, spec);
  }
  // 4) frame recurrence: one thread per spectral chain
  scan_kernel<<<(BB * CC * KINV + 255) / 256, 256, 0, stream>>>(spec, transfer, yspec);
  // 5) inverse DFT GEMM (Hann + 1/N folded into Bi)
  {
    int blocks = (BB * CC * NF / 64) * (WIN / 64);     // 8192
    inv_gemm<<<blocks, 128, 0, stream>>>(yspec, Bi, fw);
  }
  // 6) overlap-add + gain + tanh
  {
    long n = (long)BB * CC * TT;
    ola_kernel<<<(unsigned)((n + 255) / 256), 256, 0, stream>>>(fw, gain, out);
  }
}
